// BertSelfAttentionWithRelationsTableformer_65446711657070
// MI455X (gfx1250) — matrix operand order repaired and checked by reading, not compiled, not run on hardware
//
#include <hip/hip_runtime.h>

typedef __bf16 bf16;
typedef bf16  v16bf  __attribute__((ext_vector_type(16)));
typedef bf16  bf16x8 __attribute__((ext_vector_type(8)));
typedef bf16  bf16x4 __attribute__((ext_vector_type(4)));
typedef float v8f    __attribute__((ext_vector_type(8)));
typedef int   v4i    __attribute__((ext_vector_type(4)));

static constexpr int B = 8, S = 1024, D = 1024, H = 16, HD = 64, NUM_REL = 7;

union Frag16 { v16bf v; bf16x8 h[2]; };

__device__ __forceinline__ v16bf ld_frag(const bf16* p0, const bf16* p1) {
    Frag16 f;
    f.h[0] = *reinterpret_cast<const bf16x8*>(p0);
    f.h[1] = *reinterpret_cast<const bf16x8*>(p1);
    return f.v;
}

__device__ __forceinline__ v8f wmma_bf16(v16bf a, v16bf b, v8f c) {
    return __builtin_amdgcn_wmma_f32_16x16x32_bf16(false, a, false, b, (short)0, c, false, false);
}

// ---- CDNA5 async global->LDS (ASYNCcnt) with sync fallback --------------------
#if __has_builtin(__builtin_amdgcn_global_load_async_to_lds_b128)
#define HAVE_ASYNC 1
#else
#define HAVE_ASYNC 0
#endif

#if __has_builtin(__builtin_amdgcn_s_wait_asynccnt)
#define WAIT_ASYNC(n) __builtin_amdgcn_s_wait_asynccnt(n)
#else
#define WAIT_ASYNC(n) asm volatile("s_wait_asynccnt %0" ::"i"(n) : "memory")
#endif

__device__ __forceinline__ void cp16(const bf16* g, bf16* l) {
#if HAVE_ASYNC
    __builtin_amdgcn_global_load_async_to_lds_b128(
        (__attribute__((address_space(1))) v4i*)g,
        (__attribute__((address_space(3))) v4i*)l, 0, 0);
#else
    *reinterpret_cast<uint4*>(l) = *reinterpret_cast<const uint4*>(g);
#endif
}

// ---------------------------------------------------------------- cast f32 -> bf16
__global__ __launch_bounds__(256) void cast_kernel(const float* __restrict__ src,
                                                   bf16* __restrict__ dst, int n) {
    int i = (blockIdx.x * blockDim.x + threadIdx.x) * 4;
    if (i < n) {
        float4 f = *reinterpret_cast<const float4*>(src + i);
        bf16x4 o;
        o[0] = (bf16)f.x; o[1] = (bf16)f.y; o[2] = (bf16)f.z; o[3] = (bf16)f.w;
        *reinterpret_cast<bf16x4*>(dst + i) = o;
    }
}

// ---------------------------------------------------------------- fused QKV projection
// out = X @ W^T + bias, stored head-major bf16 [B,H,S,HD]; Q gets *0.125 fold-in.
// Block tile 128x128, BK=32; double-buffered LDS filled by async copies.
static constexpr int BM = 128, BN = 128, BK = 32, LPAD = 8;

__global__ __launch_bounds__(256) void qkv_gemm(
    const bf16* __restrict__ X, const bf16* __restrict__ Wall,
    const float* __restrict__ bq, const float* __restrict__ bk, const float* __restrict__ bv,
    bf16* __restrict__ Qo, bf16* __restrict__ Ko, bf16* __restrict__ Vo) {

    __shared__ bf16 As[2][BM][BK + LPAD];   // row stride 40 halves = 80B (16B multiple)
    __shared__ bf16 Bs[2][BN][BK + LPAD];

    const int z = blockIdx.z;
    const bf16* Wz = Wall + (size_t)z * D * D;
    const float* bias = (z == 0) ? bq : ((z == 1) ? bk : bv);
    bf16* out = (z == 0) ? Qo : ((z == 1) ? Ko : Vo);
    const float oscale = (z == 0) ? 0.125f : 1.0f;   // 1/sqrt(HD) folded into Q

    const int bm = blockIdx.x * BM;
    const int bn = blockIdx.y * BN;
    const int tid = threadIdx.x;
    const int wave = tid >> 5, lane = tid & 31;
    const int half = lane >> 4, l15 = lane & 15;
    const int wm = (wave & 3) * 32;   // 4 waves along M
    const int wn = (wave >> 2) * 64;  // 2 waves along N

    v8f C[2][4];
#pragma unroll
    for (int i = 0; i < 2; i++)
#pragma unroll
        for (int j = 0; j < 4; j++) C[i][j] = (v8f){0.f,0.f,0.f,0.f,0.f,0.f,0.f,0.f};

    const int lrow = tid >> 1;            // 0..127
    const int lcol = (tid & 1) * 16;      // 0 or 16 halves

    const bf16* gx0 = X  + (size_t)(bm + lrow) * D + lcol;
    const bf16* gw0 = Wz + (size_t)(bn + lrow) * D + lcol;

    auto fill = [&](int buf, int k0) {    // 4 async b128 per thread per tile pair
        cp16(gx0 + k0,     &As[buf][lrow][lcol]);
        cp16(gx0 + k0 + 8, &As[buf][lrow][lcol + 8]);
        cp16(gw0 + k0,     &Bs[buf][lrow][lcol]);
        cp16(gw0 + k0 + 8, &Bs[buf][lrow][lcol + 8]);
    };

    constexpr int NK = D / BK;            // 32 k-steps
    fill(0, 0);

    for (int kb = 0; kb < NK; kb++) {
        const int buf = kb & 1;
        if (kb + 1 < NK) {
            fill(buf ^ 1, (kb + 1) * BK); // prefetch next tile into other buffer
            WAIT_ASYNC(4);                // in-order ASYNCcnt: current buffer resident
        } else {
            WAIT_ASYNC(0);
        }
        __syncthreads();

        v16bf a[2], bfr[4];
#pragma unroll
        for (int i = 0; i < 2; i++) {     // A 16x32: runs at kbase and kbase+16
            const bf16* p = &As[buf][wm + i * 16 + l15][half * 8];
            a[i] = ld_frag(p, p + 16);
        }
#pragma unroll
        for (int j = 0; j < 4; j++) {     // B 32x16: runs at kbase and kbase+16
            const bf16* p = &Bs[buf][wn + j * 16 + l15][half * 8];
            bfr[j] = ld_frag(p, p + 16);
        }
#pragma unroll
        for (int i = 0; i < 2; i++)
#pragma unroll
            for (int j = 0; j < 4; j++) C[i][j] = wmma_bf16(a[i], bfr[j], C[i][j]);

        __syncthreads();                  // done reading before next overwrite
    }

    // epilogue: bias add, scale, scatter to [B,H,S,HD] bf16
#pragma unroll
    for (int j = 0; j < 4; j++) {
        const int ng = bn + wn + j * 16 + l15;
        const float bn_ = bias[ng];
        const int hh = ng >> 6, dd = ng & 63;
#pragma unroll
        for (int i = 0; i < 2; i++) {
#pragma unroll
            for (int r = 0; r < 8; r++) {
                const int mg = bm + wm + i * 16 + r + half * 8;
                const int bb = mg >> 10, ss = mg & 1023;
                const float val = (C[i][j][r] + bn_) * oscale;
                out[(((size_t)(bb * H + hh) * S + ss) << 6) + dd] = (bf16)val;
            }
        }
    }
}

// ---------------------------------------------------------------- flash attention w/ relation bias
__global__ __launch_bounds__(128) void attn_kernel(
    const bf16* __restrict__ Q, const bf16* __restrict__ Kx, const bf16* __restrict__ V,
    const int* __restrict__ relation, const float* __restrict__ amask,
    const float* __restrict__ rel_emb, float* __restrict__ out) {

    __shared__ bf16  Qs[64][HD + 8];    // stride 144B
    __shared__ bf16  Ks[32][HD + 8];
    __shared__ bf16  Vts[HD][32 + 8];   // V transposed: [d][key], stride 80B
    __shared__ bf16  Ps[4][16][32 + 8]; // per-wave P relayout scratch
    __shared__ float mrow[S];
    __shared__ float relt[8];

    const int qblk = blockIdx.x;
    const int bh   = blockIdx.y;
    const int bb = bh >> 4, h = bh & 15;
    const int tid = threadIdx.x, wave = tid >> 5, lane = tid & 31;
    const int half = lane >> 4, l15 = lane & 15;
    const size_t baseBH = (size_t)bh * S * HD;

    if (tid < NUM_REL) relt[tid] = rel_emb[tid * H + h];
    for (int i = tid; i < S; i += 128) mrow[i] = amask[bb * S + i];
    {   // Q tile 64x64 halves via async copies
        const int r = tid >> 1, off = (tid & 1) * 32;
        const bf16* qp = Q + baseBH + (size_t)(qblk * 64 + r) * HD + off;
#pragma unroll
        for (int u = 0; u < 4; u++) cp16(qp + u * 8, &Qs[r][off + u * 8]);
        WAIT_ASYNC(0);
    }
    __syncthreads();

    v16bf aQ[2];
    {
        const bf16* qrow = &Qs[wave * 16 + l15][0];
        const int kb8 = half * 8;
        aQ[0] = ld_frag(qrow + kb8,      qrow + kb8 + 16);
        aQ[1] = ld_frag(qrow + 32 + kb8, qrow + 32 + kb8 + 16);
    }

    v8f O[4];
#pragma unroll
    for (int j = 0; j < 4; j++) O[j] = (v8f){0.f,0.f,0.f,0.f,0.f,0.f,0.f,0.f};
    float mstat[8], lstat[8];
#pragma unroll
    for (int r = 0; r < 8; r++) { mstat[r] = -3.0e38f; lstat[r] = 0.f; }

    for (int kb = 0; kb < S / 32; kb++) {
        __syncthreads();
        {   // K tile 32x64 via async copies
            const int row = tid >> 2, off = (tid & 3) * 16;
            const bf16* kp = Kx + baseBH + (size_t)(kb * 32 + row) * HD + off;
            cp16(kp,     &Ks[row][off]);
            cp16(kp + 8, &Ks[row][off + 8]);
        }
        {   // V tile transposed into Vts[d][key] (reshape -> manual path)
            const int key = tid >> 2, doff = (tid & 3) * 16;
            const bf16* vp = V + baseBH + (size_t)(kb * 32 + key) * HD + doff;
            Frag16 t;
            t.h[0] = *reinterpret_cast<const bf16x8*>(vp);
            t.h[1] = *reinterpret_cast<const bf16x8*>(vp + 8);
#pragma unroll
            for (int j = 0; j < 16; j++) Vts[doff + j][key] = t.v[j];
        }
        WAIT_ASYNC(0);
        __syncthreads();

        // scores: two 16x16 key tiles, each summed over d via 2 WMMA steps
        v8f C0 = (v8f){0.f,0.f,0.f,0.f,0.f,0.f,0.f,0.f};
        v8f C1 = C0;
#pragma unroll
        for (int s = 0; s < 2; s++) {
            const int ko = s * 32 + half * 8;
            v16bf b0 = ld_frag(&Ks[l15][ko],      &Ks[l15][ko + 16]);
            v16bf b1 = ld_frag(&Ks[16 + l15][ko], &Ks[16 + l15][ko + 16]);
            C0 = wmma_bf16(aQ[s], b0, C0);
            C1 = wmma_bf16(aQ[s], b1, C1);
        }

        // relation bias + mask
#pragma unroll
        for (int r = 0; r < 8; r++) {
            const int qg = qblk * 64 + wave * 16 + r + half * 8;
            const int* relp = relation + ((size_t)bb * S + qg) * S + kb * 32;
            C0[r] += relt[relp[l15]]      + mrow[kb * 32 + l15];
            C1[r] += relt[relp[16 + l15]] + mrow[kb * 32 + 16 + l15];
        }

        // online softmax update
#pragma unroll
        for (int r = 0; r < 8; r++) {
            float mx = fmaxf(C0[r], C1[r]);
#pragma unroll
            for (int off = 8; off >= 1; off >>= 1) mx = fmaxf(mx, __shfl_xor(mx, off, 32));
            const float mnew  = fmaxf(mstat[r], mx);
            const float scale = __expf(mstat[r] - mnew);
            const float p0 = __expf(C0[r] - mnew);
            const float p1 = __expf(C1[r] - mnew);
            float rs = p0 + p1;
#pragma unroll
            for (int off = 8; off >= 1; off >>= 1) rs += __shfl_xor(rs, off, 32);
            lstat[r] = lstat[r] * scale + rs;
            mstat[r] = mnew;
#pragma unroll
            for (int j = 0; j < 4; j++) O[j] *= scale;
            Ps[wave][r + half * 8][l15]      = (bf16)p0;   // C-layout -> A-layout via LDS
            Ps[wave][r + half * 8][16 + l15] = (bf16)p1;
        }
        asm volatile("s_wait_dscnt 0x0" ::: "memory");     // wave-local LDS RAW fence

        {
            const bf16* pr = &Ps[wave][l15][half * 8];
            v16bf aP = ld_frag(pr, pr + 16);
#pragma unroll
            for (int j = 0; j < 4; j++) {
                const bf16* vb = &Vts[j * 16 + l15][half * 8];
                v16bf bV = ld_frag(vb, vb + 16);
                O[j] = wmma_bf16(aP, bV, O[j]);
            }
        }
    }

    // normalize and write f32 output [B,S,D]
#pragma unroll
    for (int r = 0; r < 8; r++) {
        const float inv = 1.0f / lstat[r];
        const int qg = qblk * 64 + wave * 16 + r + half * 8;
        float* op = out + ((size_t)bb * S + qg) * D + h * HD;
#pragma unroll
        for (int j = 0; j < 4; j++) op[j * 16 + l15] = O[j][r] * inv;
    }
}

// ---------------------------------------------------------------- launch
extern "C" void kernel_launch(void* const* d_in, const int* in_sizes, int n_in,
                              void* d_out, int out_size, void* d_ws, size_t ws_size,
                              hipStream_t stream) {
    const float* hidden   = (const float*)d_in[0];
    const float* amask    = (const float*)d_in[1];
    const int*   relation = (const int*)d_in[2];
    const float* Wq = (const float*)d_in[3];
    const float* bq = (const float*)d_in[4];
    const float* Wk = (const float*)d_in[5];
    const float* bk = (const float*)d_in[6];
    const float* Wv = (const float*)d_in[7];
    const float* bv = (const float*)d_in[8];
    const float* rel_emb = (const float*)d_in[9];
    float* out = (float*)d_out;

    const size_t nX = (size_t)B * S * D;   // 8,388,608
    const size_t nW = (size_t)D * D;       // 1,048,576
    bf16* Xbf = (bf16*)d_ws;
    bf16* Wbf = Xbf + nX;
    bf16* Qbf = Wbf + 3 * nW;              // [B,H,S,HD], pre-scaled by 1/8
    bf16* Kbf = Qbf + nX;
    bf16* Vbf = Kbf + nX;                  // total ws use ~70 MiB

    cast_kernel<<<(int)(nX / 1024), 256, 0, stream>>>(hidden, Xbf, (int)nX);
    cast_kernel<<<(int)(nW / 1024), 256, 0, stream>>>(Wq, Wbf + 0 * nW, (int)nW);
    cast_kernel<<<(int)(nW / 1024), 256, 0, stream>>>(Wk, Wbf + 1 * nW, (int)nW);
    cast_kernel<<<(int)(nW / 1024), 256, 0, stream>>>(Wv, Wbf + 2 * nW, (int)nW);

    qkv_gemm<<<dim3((B * S) / BM, D / BN, 3), 256, 0, stream>>>(
        Xbf, Wbf, bq, bk, bv, Qbf, Kbf, Vbf);

    attn_kernel<<<dim3(S / 64, B * H), 128, 0, stream>>>(
        Qbf, Kbf, Vbf, relation, amask, rel_emb, out);
}
